// RPN_63101659512966
// MI455X (gfx1250) — matrix operand order repaired and checked by reading, not compiled, not run on hardware
//
#include <hip/hip_runtime.h>

// ---------------- types ----------------
typedef __attribute__((ext_vector_type(16))) __bf16   v16bf;
typedef __attribute__((ext_vector_type(8)))  float    v8f;
typedef __attribute__((ext_vector_type(4)))  unsigned v4u;

union Frag { v16bf v; v4u q[2]; };

#define NTOT   261888      // total anchors
#define MTOT   87296       // total pixels
#define KPRE   12000
#define KPOST  2000

__device__ __forceinline__ unsigned short f2bf(float f) {
  unsigned u = __float_as_uint(f);
  unsigned r = u + 0x7fffu + ((u >> 16) & 1u);
  if ((u & 0x7f800000u) == 0x7f800000u) r = u;   // inf/nan passthrough
  return (unsigned short)(r >> 16);
}

// ---------------- prep: NCHW f32 -> NHWC bf16 ----------------
__global__ void k_to_nhwc_bf16(const float* __restrict__ x, unsigned short* __restrict__ out, int HW) {
  int i = blockIdx.x * 256 + threadIdx.x;           // i over HW*256
  if (i >= HW * 256) return;
  int m = i >> 8, c = i & 255;
  out[(size_t)m * 256 + c] = f2bf(x[(size_t)c * HW + m]);
}

// ---------------- prep: conv_w (O,I,3,3) f32 -> [tap][O][I] bf16 ----------------
__global__ void k_pack_w(const float* __restrict__ w, unsigned short* __restrict__ wp) {
  int i = blockIdx.x * 256 + threadIdx.x;           // 9*256*256
  if (i >= 9 * 65536) return;
  int tap = i >> 16, rem = i & 65535;
  int n = rem >> 8, c = rem & 255;
  wp[i] = f2bf(w[(size_t)n * 2304 + c * 9 + tap]);
}

// ---------------- 3x3 conv (implicit GEMM, bf16 WMMA) + ReLU ----------------
// A-frag (16x32 bf16, ISA 7.12.2): lane = M row; lanes<16 hold K k0+0..7 / k0+16..23,
// lanes>=16 hold K k0+8..15 / k0+24..31.  B-frag (32x16): lane holds col n=lane&15,
// same K split as A.  C: VGPR r -> M = 8*(lane>>4)+r, N = lane&15.
// Padding: out-of-bounds taps read a 512B zero page (pointer select per tap) so the
// K-loop has no per-element masking and no WMMA<->VALU hazard NOPs.
__global__ __launch_bounds__(256)
void k_conv3x3_wmma(const unsigned short* __restrict__ xbf,   // [H*W][256] bf16
                    const unsigned short* __restrict__ wp,    // [9][256][256] bf16
                    const float* __restrict__ conv_b,
                    const unsigned short* __restrict__ zpage, // 256 bf16 zeros
                    unsigned short* __restrict__ hid,         // [H*W][256] bf16
                    int H, int W) {
  const int lane = threadIdx.x & 31;
  const int wave = threadIdx.x >> 5;                 // 0..7
  const int l15  = lane & 15;
  const int ksel = lane >> 4;                        // 0 or 1
  const int mt   = blockIdx.x * 16;
  const int nt   = (blockIdx.y * 8 + wave) * 16;     // 0..240
  const int row  = mt + l15;                         // this lane's A row (pixel)
  const int y = row / W, x = row - (row / W) * W;

  v8f acc0 = {}, acc1 = {};
  const unsigned short* bbase = wp + (size_t)(nt + l15) * 256;

  for (int tap = 0; tap < 9; ++tap) {
    const int dy = tap / 3 - 1, dx = tap - (tap / 3) * 3 - 1;
    const int yy = y + dy, xx = x + dx;
    const bool valid = (yy >= 0) & (yy < H) & (xx >= 0) & (xx < W);
    const unsigned short* arow =
        valid ? (xbf + (size_t)(yy * W + xx) * 256) : zpage;
    const unsigned short* brow = bbase + (size_t)tap * 65536;
    if (tap < 8) {   // prefetch next tap's A row (global_prefetch_b8)
      const int ny = y + (tap + 1) / 3 - 1, nx = x + (tap + 1) % 3 - 1;
      const bool pv = (ny >= 0) & (ny < H) & (nx >= 0) & (nx < W);
      __builtin_prefetch(pv ? (xbf + (size_t)(ny * W + nx) * 256) : zpage, 0, 1);
    }
#pragma unroll
    for (int k0 = 0; k0 < 256; k0 += 64) {
      // even K-step -> acc0
      Frag a0, b0;
      a0.q[0] = *(const v4u*)(arow + k0 + ksel * 8);
      a0.q[1] = *(const v4u*)(arow + k0 + 16 + ksel * 8);
      b0.q[0] = *(const v4u*)(brow + k0 + ksel * 8);
      b0.q[1] = *(const v4u*)(brow + k0 + 16 + ksel * 8);
      acc0 = __builtin_amdgcn_wmma_f32_16x16x32_bf16(false, a0.v, false, b0.v,
                                                     (short)0, acc0, false, false);
      // odd K-step -> acc1 (independent chain)
      Frag a1, b1;
      a1.q[0] = *(const v4u*)(arow + k0 + 32 + ksel * 8);
      a1.q[1] = *(const v4u*)(arow + k0 + 48 + ksel * 8);
      b1.q[0] = *(const v4u*)(brow + k0 + 32 + ksel * 8);
      b1.q[1] = *(const v4u*)(brow + k0 + 48 + ksel * 8);
      acc1 = __builtin_amdgcn_wmma_f32_16x16x32_bf16(false, a1.v, false, b1.v,
                                                     (short)0, acc1, false, false);
    }
  }
  const int ncol = nt + l15;
  const float bias = conv_b[ncol];
  const int m0 = mt + ksel * 8;
#pragma unroll
  for (int r = 0; r < 8; ++r) {
    float v = acc0[r] + acc1[r] + bias;
    v = v > 0.f ? v : 0.f;
    hid[(size_t)(m0 + r) * 256 + ncol] = f2bf(v);
  }
}

// ---------------- 1x1 heads: scores(6) + locs(12) per pixel ----------------
__global__ __launch_bounds__(128)
void k_heads(const unsigned short* __restrict__ hid,
             const float* __restrict__ score_w, const float* __restrict__ score_b,
             const float* __restrict__ loc_w,   const float* __restrict__ loc_b,
             float* __restrict__ out_locs, float* __restrict__ out_scores,
             float* __restrict__ scores1, int Mtot) {
  __shared__ float wsh[18 * 256];                    // rows 0..5 score, 6..17 loc
  for (int o = threadIdx.x; o < 18 * 256; o += 128)
    wsh[o] = (o < 1536) ? score_w[o] : loc_w[o - 1536];
  __syncthreads();

  int m = blockIdx.x * 128 + threadIdx.x;
  if (m >= Mtot) return;
  float acc[18];
#pragma unroll
  for (int o = 0; o < 18; ++o) acc[o] = (o < 6) ? score_b[o] : loc_b[o - 6];
  const v4u* hp4 = (const v4u*)(hid + (size_t)m * 256);   // 16B-aligned
  for (int c8 = 0; c8 < 32; ++c8) {                       // 8 bf16 per b128 load
    v4u h8 = hp4[c8];
#pragma unroll
    for (int q = 0; q < 4; ++q) {
      unsigned u = h8[q];
      float h0 = __uint_as_float(u << 16);                // low bf16
      float h1 = __uint_as_float(u & 0xffff0000u);        // high bf16
      int c = c8 * 8 + q * 2;
#pragma unroll
      for (int o = 0; o < 18; ++o)
        acc[o] += h0 * wsh[o * 256 + c] + h1 * wsh[o * 256 + c + 1];
    }
  }
  float* ol = out_locs + (size_t)m * 12;
#pragma unroll
  for (int j = 0; j < 12; ++j) ol[j] = acc[6 + j];
  float* os = out_scores + (size_t)m * 6;
#pragma unroll
  for (int j = 0; j < 6; ++j) os[j] = acc[j];
#pragma unroll
  for (int a = 0; a < 3; ++a) scores1[(size_t)m * 3 + a] = acc[2 * a + 1];  // fg logit
}

// ---------------- anchors + decode + sortable keys ----------------
__global__ void k_anchor_decode(const float* __restrict__ out_locs,
                                float* __restrict__ out_anchors,
                                float* __restrict__ boxes,
                                const float* __restrict__ scores1,
                                unsigned* __restrict__ keys) {
  int i = blockIdx.x * 256 + threadIdx.x;
  if (i >= NTOT) return;
  int m = i / 3, a = i - m * 3;
  int mloc, Wl, st;
  if      (m < 65536) { mloc = m;         Wl = 256; st = 4;  }
  else if (m < 81920) { mloc = m - 65536; Wl = 128; st = 8;  }
  else if (m < 86016) { mloc = m - 81920; Wl = 64;  st = 16; }
  else if (m < 87040) { mloc = m - 86016; Wl = 32;  st = 32; }
  else                { mloc = m - 87040; Wl = 16;  st = 64; }
  int y = mloc / Wl, x = mloc - y * Wl;
  float sr  = (a == 0) ? 0.70710678118654752f : ((a == 1) ? 1.f : 1.41421356237309505f);
  float sri = (a == 0) ? 1.41421356237309505f : ((a == 1) ? 1.f : 0.70710678118654752f);
  float fs = (float)st;
  float h = fs * 8.f * sr, w = fs * 8.f * sri;
  float cy = y * fs + 0.5f * fs, cx = x * fs + 0.5f * fs;
  out_anchors[(size_t)i * 4 + 0] = cy - 0.5f * h;
  out_anchors[(size_t)i * 4 + 1] = cx - 0.5f * w;
  out_anchors[(size_t)i * 4 + 2] = cy + 0.5f * h;
  out_anchors[(size_t)i * 4 + 3] = cx + 0.5f * w;
  float l0 = out_locs[(size_t)i * 4 + 0], l1 = out_locs[(size_t)i * 4 + 1];
  float l2 = out_locs[(size_t)i * 4 + 2], l3 = out_locs[(size_t)i * 4 + 3];
  float ncy = l0 * h + cy, ncx = l1 * w + cx;
  float nh = expf(l2) * h, nw = expf(l3) * w;
  boxes[(size_t)i * 4 + 0] = ncy - 0.5f * nh;
  boxes[(size_t)i * 4 + 1] = ncx - 0.5f * nw;
  boxes[(size_t)i * 4 + 2] = ncy + 0.5f * nh;
  boxes[(size_t)i * 4 + 3] = ncx + 0.5f * nw;
  unsigned u = __float_as_uint(scores1[i]);
  keys[i] = (u & 0x80000000u) ? ~u : (u | 0x80000000u);   // monotone float key
}

// ---------------- exact top-K (two-pass 16-bit radix select) ----------------
__global__ void k_hist_hi(const unsigned* __restrict__ keys, unsigned* __restrict__ hist) {
  int i = blockIdx.x * 256 + threadIdx.x;
  if (i < NTOT) atomicAdd(&hist[keys[i] >> 16], 1u);
}
__global__ void k_scan_hi(const unsigned* __restrict__ hist, unsigned* __restrict__ sel) {
  if (threadIdx.x != 0) return;
  unsigned cum = 0;
  for (int b = 65535; b >= 0; --b) {
    unsigned c = hist[b];
    if (cum + c >= (unsigned)KPRE) { sel[2] = (unsigned)b; sel[3] = cum; return; }
    cum += c;
  }
}
__global__ void k_hist_lo(const unsigned* __restrict__ keys, const unsigned* __restrict__ sel,
                          unsigned* __restrict__ hist) {
  int i = blockIdx.x * 256 + threadIdx.x;
  if (i < NTOT && (keys[i] >> 16) == sel[2]) atomicAdd(&hist[keys[i] & 0xffffu], 1u);
}
__global__ void k_scan_lo(const unsigned* __restrict__ hist, unsigned* __restrict__ sel) {
  if (threadIdx.x != 0) return;
  unsigned cum = sel[3];
  for (int b = 65535; b >= 0; --b) {
    unsigned c = hist[b];
    if (cum + c >= (unsigned)KPRE) {
      sel[4] = (sel[2] << 16) | (unsigned)b;   // threshold key T
      sel[5] = cum;                            // count strictly above T
      sel[6] = (unsigned)KPRE - cum;           // ties to take at T
      return;
    }
    cum += c;
  }
}
__global__ void k_compact(const unsigned* __restrict__ keys, const float* __restrict__ boxes,
                          const float* __restrict__ scores1, const unsigned* __restrict__ sel,
                          unsigned* __restrict__ ctr, float* __restrict__ cbox,
                          float* __restrict__ cscore, float* __restrict__ carea,
                          const int* __restrict__ ph, const int* __restrict__ pw) {
  int i = blockIdx.x * 256 + threadIdx.x;
  if (i >= NTOT) return;
  unsigned k = keys[i], T = sel[4], G = sel[5], t = sel[6];
  int slot = -1;
  if (k > T) slot = (int)atomicAdd(&ctr[0], 1u);
  else if (k == T) {
    unsigned e = atomicAdd(&ctr[1], 1u);
    if (e < t) slot = (int)(G + e);
  }
  if (slot < 0) return;
  float ih = (float)ph[0], iw = (float)pw[0];
  float y1 = fminf(fmaxf(boxes[(size_t)i * 4 + 0], 0.f), ih);
  float x1 = fminf(fmaxf(boxes[(size_t)i * 4 + 1], 0.f), iw);
  float y2 = fminf(fmaxf(boxes[(size_t)i * 4 + 2], 0.f), ih);
  float x2 = fminf(fmaxf(boxes[(size_t)i * 4 + 3], 0.f), iw);
  cbox[(size_t)slot * 4 + 0] = y1; cbox[(size_t)slot * 4 + 1] = x1;
  cbox[(size_t)slot * 4 + 2] = y2; cbox[(size_t)slot * 4 + 3] = x2;
  float hh = y2 - y1, ww = x2 - x1;
  cscore[slot] = (hh >= 16.f && ww >= 16.f) ? scores1[i] : -__builtin_inff();
  carea[slot] = hh * ww;
}

// ---------------- greedy NMS, single workgroup ----------------
__global__ __launch_bounds__(1024)
void k_nms(const float* __restrict__ cbox, const float* __restrict__ cscore,
           const float* __restrict__ carea, float* __restrict__ rois) {
  __shared__ float sc[KPRE];          // 48 KB
  __shared__ float redv[1024];
  __shared__ int   redi[1024];
  __shared__ float bb[5];
  const int t = threadIdx.x;
  for (int j = t; j < KPRE; j += 1024) sc[j] = cscore[j];
  __syncthreads();
  for (int it = 0; it < KPOST; ++it) {
    float best = -__builtin_inff(); int bi = 0;
    for (int j = t; j < KPRE; j += 1024) {
      float v = sc[j];
      if (v > best) { best = v; bi = j; }
    }
    redv[t] = best; redi[t] = bi;
    __syncthreads();
    for (int s = 512; s > 0; s >>= 1) {
      if (t < s) {
        float v2 = redv[t + s]; int i2 = redi[t + s];
        if (v2 > redv[t] || (v2 == redv[t] && i2 < redi[t])) { redv[t] = v2; redi[t] = i2; }
      }
      __syncthreads();
    }
    const int widx = redi[0];
    const bool ok = redv[0] > -__builtin_inff();
    if (t == 0) {
      float b0 = cbox[(size_t)widx * 4 + 0], b1 = cbox[(size_t)widx * 4 + 1];
      float b2 = cbox[(size_t)widx * 4 + 2], b3 = cbox[(size_t)widx * 4 + 3];
      rois[(size_t)it * 4 + 0] = ok ? b0 : 0.f;
      rois[(size_t)it * 4 + 1] = ok ? b1 : 0.f;
      rois[(size_t)it * 4 + 2] = ok ? b2 : 0.f;
      rois[(size_t)it * 4 + 3] = ok ? b3 : 0.f;
      bb[0] = b0; bb[1] = b1; bb[2] = b2; bb[3] = b3; bb[4] = carea[widx];
      sc[widx] = -__builtin_inff();
    }
    __syncthreads();
    if (ok) {
      float by1 = bb[0], bx1 = bb[1], by2 = bb[2], bx2 = bb[3], ba = bb[4];
      for (int j = t; j < KPRE; j += 1024) {
        float y1 = fmaxf(by1, cbox[(size_t)j * 4 + 0]);
        float x1 = fmaxf(bx1, cbox[(size_t)j * 4 + 1]);
        float y2 = fminf(by2, cbox[(size_t)j * 4 + 2]);
        float x2 = fminf(bx2, cbox[(size_t)j * 4 + 3]);
        float inter = fmaxf(y2 - y1, 0.f) * fmaxf(x2 - x1, 0.f);
        float iou = inter / (carea[j] + ba - inter + 1e-10f);
        if (iou > 0.7f) sc[j] = -__builtin_inff();
      }
    }
    __syncthreads();
  }
}

// ---------------- host ----------------
extern "C" void kernel_launch(void* const* d_in, const int* in_sizes, int n_in,
                              void* d_out, int out_size, void* d_ws, size_t ws_size,
                              hipStream_t stream) {
  const float* feat[5] = {(const float*)d_in[0], (const float*)d_in[1], (const float*)d_in[2],
                          (const float*)d_in[3], (const float*)d_in[4]};
  const float* conv_w  = (const float*)d_in[5];
  const float* conv_b  = (const float*)d_in[6];
  const float* score_w = (const float*)d_in[7];
  const float* score_b = (const float*)d_in[8];
  const float* loc_w   = (const float*)d_in[9];
  const float* loc_b   = (const float*)d_in[10];
  const int*   ph      = (const int*)d_in[11];
  const int*   pw      = (const int*)d_in[12];

  float* out_locs    = (float*)d_out;                 // N*4
  float* out_scores  = out_locs + (size_t)NTOT * 4;   // N*2
  float* out_rois    = out_scores + (size_t)NTOT * 2; // 2000*4
  float* out_anchors = out_rois + (size_t)KPOST * 4;  // N*4

  char* ws = (char*)d_ws;
  const size_t XBF_BYTES = (size_t)MTOT * 256 * 2;    // 44,695,552
  unsigned short* xbf   = (unsigned short*)ws;
  unsigned short* hid   = (unsigned short*)(ws + XBF_BYTES);
  unsigned short* wp    = (unsigned short*)(ws + 2 * XBF_BYTES);
  unsigned short* zpage = (unsigned short*)(ws + 2 * XBF_BYTES + 1179648); // 512B zeros
  // overlay into xbf region (only used after all convs finish):
  float*    boxes   = (float*)ws;                          // N*4 f32
  float*    scores1 = (float*)(ws + 4190208);              // N f32
  unsigned* keys    = (unsigned*)(ws + 5237760);           // N u32
  unsigned* hist    = (unsigned*)(ws + 6285312);           // 65536 u32
  float*    cbox    = (float*)(ws + 6547456);              // 12000*4
  float*    cscore  = (float*)(ws + 6739456);              // 12000
  float*    carea   = (float*)(ws + 6787456);              // 12000
  unsigned* selctr  = (unsigned*)(ws + 6835456);           // [0..1]=ctr, [2..6]=sel

  static const int    Hs[5]   = {256, 128, 64, 32, 16};
  static const size_t Poff[5] = {0, 65536, 81920, 86016, 87040};

  for (int l = 0; l < 5; ++l) {
    int HW = Hs[l] * Hs[l];
    k_to_nhwc_bf16<<<(HW * 256 + 255) / 256, 256, 0, stream>>>(feat[l], xbf + Poff[l] * 256, HW);
  }
  k_pack_w<<<(9 * 65536 + 255) / 256, 256, 0, stream>>>(conv_w, wp);
  hipMemsetAsync(zpage, 0, 512, stream);
  for (int l = 0; l < 5; ++l) {
    int HW = Hs[l] * Hs[l];
    dim3 grid(HW / 16, 2);
    k_conv3x3_wmma<<<grid, 256, 0, stream>>>(xbf + Poff[l] * 256, wp, conv_b, zpage,
                                             hid + Poff[l] * 256, Hs[l], Hs[l]);
  }
  k_heads<<<(MTOT + 127) / 128, 128, 0, stream>>>(hid, score_w, score_b, loc_w, loc_b,
                                                  out_locs, out_scores, scores1, MTOT);
  k_anchor_decode<<<(NTOT + 255) / 256, 256, 0, stream>>>(out_locs, out_anchors, boxes,
                                                          scores1, keys);
  hipMemsetAsync(hist, 0, 65536 * sizeof(unsigned), stream);
  hipMemsetAsync(selctr, 0, 8 * sizeof(unsigned), stream);
  k_hist_hi<<<(NTOT + 255) / 256, 256, 0, stream>>>(keys, hist);
  k_scan_hi<<<1, 32, 0, stream>>>(hist, selctr);
  hipMemsetAsync(hist, 0, 65536 * sizeof(unsigned), stream);
  k_hist_lo<<<(NTOT + 255) / 256, 256, 0, stream>>>(keys, selctr, hist);
  k_scan_lo<<<1, 32, 0, stream>>>(hist, selctr);
  k_compact<<<(NTOT + 255) / 256, 256, 0, stream>>>(keys, boxes, scores1, selctr, selctr,
                                                    cbox, cscore, carea, ph, pw);
  k_nms<<<1, 1024, 0, stream>>>(cbox, cscore, carea, out_rois);
}
